// MSAColumnAttention_18176301597137
// MI455X (gfx1250) — matrix operand order repaired and checked by reading, not compiled, not run on hardware
//
#include <hip/hip_runtime.h>
#include <hip/hip_bf16.h>

// ---------------------------------------------------------------------------
// MSA column attention for MI455X (gfx1250), bf16 WMMA everywhere.
//   N_SEQ=256, N_RES=384, C_M=256, H=8, D=32
// Workspace layout (bf16 elements):
//   [0)        WqT[256][256]   (query_w^T * 1/sqrt(32))
//   [65536)    WkT[256][256]
//   [131072)   WvT[256][256]
//   [196608)   WgT[256][256]
//   [262144)   WoT[256][256]   (output_w^T, [o][h*32+c])
//   [327680)   xn [384][256][256]  LayerNormed activations, bf16
// Total ws: (327680 + 384*256*256) * 2 bytes ~= 48.6 MB
// ---------------------------------------------------------------------------

typedef __attribute__((ext_vector_type(16))) __bf16 v16bf;
typedef __attribute__((ext_vector_type(8)))  __bf16 v8bf;
typedef __attribute__((ext_vector_type(8)))  float  v8f;

#define N_SEQ 256
#define N_RES 384
#define C_M   256
#define N_HEAD 8
#define HEAD_DIM 32

#if __has_builtin(__builtin_amdgcn_global_load_async_to_lds_b32) && \
    __has_builtin(__builtin_amdgcn_s_wait_asynccnt)
#define HAVE_ASYNC_LDS 1
typedef __attribute__((address_space(1))) int GASInt;   // global addrspace
typedef __attribute__((address_space(3))) int LASInt;   // LDS addrspace
#endif

// ---- WMMA fragment helpers (wave32; K contiguous per lane half, same order in A and B) ----------
// A (16x32 bf16): lane L holds row M = L%16; 16 contiguous K starting at 16*(L/16) (matches frag_b)
__device__ __forceinline__ v16bf frag_a(const __bf16* rowbase, int lane) {
    return *(const v16bf*)(rowbase + ((lane >> 4) << 4));
}
// B (32x16 bf16): lane L holds column N = L%16; 16 contiguous K starting at 16*(L/16).
// `colbase` must point at K=0 of the column (i.e. B stored transposed, [n][k]).
__device__ __forceinline__ v16bf frag_b(const __bf16* colbase, int lane) {
    return *(const v16bf*)(colbase + ((lane >> 4) << 4));
}
__device__ __forceinline__ v8f wmma_bf16(v16bf a, v16bf b, v8f c) {
    return __builtin_amdgcn_wmma_f32_16x16x32_bf16(false, a, false, b, (short)0, c, false, false);
}

// ---------------------------------------------------------------------------
// Kernel 1: transpose + convert the 5 weight matrices to bf16.
// dst is 5 contiguous [256][256] mats; dst[w][n][k] = src[w][k][n] (*qscale for w==0)
// ---------------------------------------------------------------------------
__global__ __launch_bounds__(256) void prep_weights(
    const float* __restrict__ qw, const float* __restrict__ kw,
    const float* __restrict__ vw, const float* __restrict__ gw,
    const float* __restrict__ ow, __bf16* __restrict__ dst)
{
    int id  = blockIdx.x * 256 + threadIdx.x;   // < 5*65536
    int wsel = id >> 16;
    int rem  = id & 65535;
    int n = rem >> 8, k = rem & 255;
    const float* src = (wsel == 0) ? qw : (wsel == 1) ? kw :
                       (wsel == 2) ? vw : (wsel == 3) ? gw : ow;
    float val = src[k * 256 + n];
    if (wsel == 0) val *= 0.17677669529663687f;   // 1/sqrt(HEAD_DIM)
    dst[id] = (__bf16)val;
}

// ---------------------------------------------------------------------------
// Kernel 2: LayerNorm over channels, write bf16 xn[b][s][c].
// One wave (32 lanes) per (s,b) row; lane owns 8 contiguous channels.
// ---------------------------------------------------------------------------
__global__ __launch_bounds__(256) void ln_to_bf16(
    const float* __restrict__ x,        // [256][384][256]
    const float* __restrict__ sc, const float* __restrict__ bi,
    __bf16* __restrict__ xn)            // [384][256][256]
{
    const int wave = threadIdx.x >> 5, lane = threadIdx.x & 31;
    const int rid  = blockIdx.x * 8 + wave;       // rid = s*384 + b
    const int s = rid / N_RES, b = rid % N_RES;
    const float* row = x + (size_t)rid * C_M;

    float4 a = ((const float4*)row)[lane * 2];
    float4 c = ((const float4*)row)[lane * 2 + 1];
    float v0[8] = {a.x, a.y, a.z, a.w, c.x, c.y, c.z, c.w};

    float s1 = 0.f, s2 = 0.f;
#pragma unroll
    for (int i = 0; i < 8; ++i) { s1 += v0[i]; s2 += v0[i] * v0[i]; }
#pragma unroll
    for (int m = 1; m < 32; m <<= 1) {
        s1 += __shfl_xor(s1, m, 32);
        s2 += __shfl_xor(s2, m, 32);
    }
    const float mu  = s1 * (1.f / 256.f);
    const float var = s2 * (1.f / 256.f) - mu * mu;
    const float inv = rsqrtf(var + 1e-5f);

    const int c0 = lane * 8;
    v8bf o;
#pragma unroll
    for (int i = 0; i < 8; ++i)
        o[i] = (__bf16)((v0[i] - mu) * inv * sc[c0 + i] + bi[c0 + i]);
    *(v8bf*)(xn + ((size_t)b * N_SEQ + s) * C_M + c0) = o;
}

// ---------------------------------------------------------------------------
// Kernel 3: per-column fused attention. grid.x = 384 columns, 256 threads.
// Dynamic LDS (289 KB):
//   Qh[256][32] Kh[256][32] Vt[32][256] Gh[256][32]  bf16  (64 KB)
//   AG[256][256] bf16 (gated weighted-avg accumulator)     (128 KB)
//   Sf[64][256] f32  (logits chunk)                        (64 KB)
//   Sb[64][256] bf16 (softmax weights chunk)               (32 KB)
//   Bias[256]   f32  (mask bias row, async-staged)         (1 KB)
// ---------------------------------------------------------------------------
__global__ __launch_bounds__(256) void msa_core(
    const __bf16* __restrict__ xn,   // [384][256][256]
    const __bf16* __restrict__ WqT, const __bf16* __restrict__ WkT,
    const __bf16* __restrict__ WvT, const __bf16* __restrict__ WgT,
    const __bf16* __restrict__ WoT,
    const float*  __restrict__ msa_mask,  // [256][384]
    const float*  __restrict__ gating_b,  // [8][32]
    const float*  __restrict__ output_b,  // [256]
    float* __restrict__ out)              // [256][384][256]
{
    extern __shared__ char smem_raw[];
    __bf16* Qh = (__bf16*)smem_raw;           // 256*32
    __bf16* Kh = Qh + N_SEQ * HEAD_DIM;       // 256*32
    __bf16* Vt = Kh + N_SEQ * HEAD_DIM;       // [32][256]
    __bf16* Gh = Vt + HEAD_DIM * N_SEQ;       // 256*32
    __bf16* AG = Gh + N_SEQ * HEAD_DIM;       // [256][256]
    float*  Sf = (float*)(AG + N_SEQ * C_M);  // [64][256]
    __bf16* Sb = (__bf16*)(Sf + 64 * N_SEQ);  // [64][256]
    float*  Bias = (float*)(Sb + 64 * N_SEQ); // [256]

    const int b    = blockIdx.x;
    const int tid  = threadIdx.x;
    const int w    = tid >> 5;       // wave 0..7
    const int lane = tid & 31;
    const int lh   = lane >> 4;      // half-select
    const int ln   = lane & 15;      // row/col within 16

    const __bf16* xb = xn + (size_t)b * N_SEQ * C_M;

    // ---- stage mask column -> LDS once per block (async-to-LDS path) ------
#ifdef HAVE_ASYNC_LDS
    {
        __builtin_amdgcn_global_load_async_to_lds_b32(
            (GASInt*)(msa_mask + (size_t)tid * N_RES + b),
            (LASInt*)(Bias + tid), 0, 0);
        __builtin_amdgcn_s_wait_asynccnt(0);
    }
#else
    Bias[tid] = msa_mask[(size_t)tid * N_RES + b];
#endif
    __syncthreads();
    Bias[tid] = 1e9f * (Bias[tid] - 1.0f);
    __syncthreads();

    for (int h = 0; h < N_HEAD; ++h) {
        // ---- prefetch next head's weight slices into GL2/L0 ---------------
        if (h + 1 < N_HEAD && tid < 128) {
            const size_t nb = (size_t)(h + 1) * HEAD_DIM * C_M + (size_t)tid * 64;
            __builtin_prefetch(WqT + nb, 0, 0);
            __builtin_prefetch(WkT + nb, 0, 0);
            __builtin_prefetch(WvT + nb, 0, 0);
            __builtin_prefetch(WgT + nb, 0, 0);
        }

        // ---- projections: Q,K,G -> [256][32]; V -> Vt[32][256] ------------
        const __bf16* wq = WqT + (size_t)h * HEAD_DIM * C_M;
        const __bf16* wk = WkT + (size_t)h * HEAD_DIM * C_M;
        const __bf16* wv = WvT + (size_t)h * HEAD_DIM * C_M;
        const __bf16* wg = WgT + (size_t)h * HEAD_DIM * C_M;
        for (int t = w; t < 32; t += 8) {
            const int mt = t >> 1, nt = t & 1;
            const int m = mt * 16 + ln;           // A row (seq)
            const int n = nt * 16 + ln;           // B col (0..31 within head)
            const __bf16* arow = xb + (size_t)m * C_M;
            v8f aq = {}, ak = {}, av = {}, ag = {};
#pragma unroll
            for (int kb = 0; kb < 8; ++kb) {
                v16bf a  = frag_a(arow + kb * 32, lane);
                v16bf bq = frag_b(wq + (size_t)n * C_M + kb * 32, lane);
                v16bf bk = frag_b(wk + (size_t)n * C_M + kb * 32, lane);
                v16bf bv = frag_b(wv + (size_t)n * C_M + kb * 32, lane);
                v16bf bg = frag_b(wg + (size_t)n * C_M + kb * 32, lane);
                aq = wmma_bf16(a, bq, aq);
                ak = wmma_bf16(a, bk, ak);
                av = wmma_bf16(a, bv, av);
                ag = wmma_bf16(a, bg, ag);
            }
            const float gb = gating_b[h * HEAD_DIM + n];
#pragma unroll
            for (int r = 0; r < 8; ++r) {
                const int mm = mt * 16 + lh * 8 + r;
                Qh[mm * HEAD_DIM + n] = (__bf16)aq[r];
                Kh[mm * HEAD_DIM + n] = (__bf16)ak[r];
                const float g = ag[r] + gb;
                Gh[mm * HEAD_DIM + n] = (__bf16)(1.f / (1.f + __expf(-g)));
            }
            v8bf pv;
#pragma unroll
            for (int r = 0; r < 8; ++r) pv[r] = (__bf16)av[r];
            *(v8bf*)(Vt + (size_t)n * N_SEQ + mt * 16 + lh * 8) = pv;
        }
        __syncthreads();

        // ---- attention in 4 chunks of 64 query rows -----------------------
        for (int qc = 0; qc < 4; ++qc) {
            // logits: 4x16 tiles of 16x16, one WMMA each (K = head_dim = 32)
            for (int t = w; t < 64; t += 8) {
                const int mt = t >> 4, nt = t & 15;
                const int m    = qc * 64 + mt * 16 + ln;   // q row
                const int kseq = nt * 16 + ln;             // k col
                v16bf a  = frag_a(Qh + (size_t)m * HEAD_DIM, lane);
                v16bf bb = frag_b(Kh + (size_t)kseq * HEAD_DIM, lane);
                v8f acc = {};
                acc = wmma_bf16(a, bb, acc);
                const float bias = Bias[kseq];
#pragma unroll
                for (int r = 0; r < 8; ++r)
                    Sf[(mt * 16 + lh * 8 + r) * N_SEQ + kseq] = acc[r] + bias;
            }
            __syncthreads();

            // softmax: 4 lanes per row, 64 cols each; two-pass expf
            {
                const int row = tid >> 2, sub = tid & 3;
                const float* rp = Sf + (size_t)row * N_SEQ + sub * 64;
                float mx = -3.0e38f;
#pragma unroll 8
                for (int i = 0; i < 64; ++i) mx = fmaxf(mx, rp[i]);
                mx = fmaxf(mx, __shfl_xor(mx, 1, 32));
                mx = fmaxf(mx, __shfl_xor(mx, 2, 32));
                float sm = 0.f;
#pragma unroll 8
                for (int i = 0; i < 64; ++i) sm += __expf(rp[i] - mx);
                sm += __shfl_xor(sm, 1, 32);
                sm += __shfl_xor(sm, 2, 32);
                const float inv = 1.0f / sm;
                __bf16* op = Sb + (size_t)row * N_SEQ + sub * 64;
#pragma unroll 8
                for (int i = 0; i < 64; ++i)
                    op[i] = (__bf16)(__expf(rp[i] - mx) * inv);
            }
            __syncthreads();

            // PV: [64 x 256] x [256 x 32] -> 8 tiles, one per wave
            {
                const int mt = w >> 1, nt = w & 1;
                const int mrel = mt * 16 + ln;
                const int c    = nt * 16 + ln;
                v8f acc = {};
#pragma unroll
                for (int kb = 0; kb < 8; ++kb) {
                    v16bf a  = frag_a(Sb + (size_t)mrel * N_SEQ + kb * 32, lane);
                    v16bf bb = frag_b(Vt + (size_t)c * N_SEQ + kb * 32, lane);
                    acc = wmma_bf16(a, bb, acc);
                }
#pragma unroll
                for (int r = 0; r < 8; ++r) {
                    const int m = qc * 64 + mt * 16 + lh * 8 + r;
                    const float g = (float)Gh[m * HEAD_DIM + c];
                    AG[(size_t)m * C_M + h * HEAD_DIM + c] = (__bf16)(acc[r] * g);
                }
            }
            __syncthreads();
        }
        __syncthreads();   // Qh/Kh/Vt/Gh reused next head
    }

    // ---- output projection: AG[256][256] @ WoT^T + bias -> out[:, b, :] ---
    for (int t = w; t < 256; t += 8) {
        const int mt = t >> 4, nt = t & 15;
        const int m = mt * 16 + ln;
        const int o = nt * 16 + ln;
        v8f acc = {};
#pragma unroll
        for (int kb = 0; kb < 8; ++kb) {
            v16bf a  = frag_a(AG + (size_t)m * C_M + kb * 32, lane);
            v16bf bb = frag_b(WoT + (size_t)o * C_M + kb * 32, lane);
            acc = wmma_bf16(a, bb, acc);
        }
        const float ob = output_b[o];
#pragma unroll
        for (int r = 0; r < 8; ++r) {
            const int s = mt * 16 + lh * 8 + r;
            out[((size_t)s * N_RES + b) * C_M + o] = acc[r] + ob;
        }
    }
}

// ---------------------------------------------------------------------------
extern "C" void kernel_launch(void* const* d_in, const int* in_sizes, int n_in,
                              void* d_out, int out_size, void* d_ws, size_t ws_size,
                              hipStream_t stream) {
    const float* msa_act  = (const float*)d_in[0];
    const float* msa_mask = (const float*)d_in[1];
    const float* ln_scale = (const float*)d_in[2];
    const float* ln_bias  = (const float*)d_in[3];
    const float* query_w  = (const float*)d_in[4];
    const float* key_w    = (const float*)d_in[5];
    const float* value_w  = (const float*)d_in[6];
    const float* gating_w = (const float*)d_in[7];
    const float* gating_b = (const float*)d_in[8];
    const float* output_w = (const float*)d_in[9];
    const float* output_b = (const float*)d_in[10];
    float* out = (float*)d_out;

    __bf16* ws  = (__bf16*)d_ws;
    __bf16* WqT = ws;
    __bf16* WkT = WqT + 65536;
    __bf16* WvT = WkT + 65536;
    __bf16* WgT = WvT + 65536;
    __bf16* WoT = WgT + 65536;
    __bf16* xn  = WoT + 65536;   // [384][256][256]

    // 1) weights -> transposed bf16 (5 * 65536 elements)
    prep_weights<<<(5 * 65536) / 256, 256, 0, stream>>>(
        query_w, key_w, value_w, gating_w, output_w, WqT);

    // 2) LayerNorm -> bf16 staging (98304 rows, 8 rows/block)
    ln_to_bf16<<<(N_SEQ * N_RES) / 8, 256, 0, stream>>>(
        msa_act, ln_scale, ln_bias, xn);

    // 3) fused per-column attention
    constexpr size_t SMEM =
        (size_t)(4 * N_SEQ * HEAD_DIM + N_SEQ * C_M) * sizeof(__bf16)  // Qh,Kh,Vt,Gh,AG
        + (size_t)64 * N_SEQ * sizeof(float)                            // Sf
        + (size_t)64 * N_SEQ * sizeof(__bf16)                           // Sb
        + (size_t)N_SEQ * sizeof(float);                                // Bias = 289 KB
    (void)hipFuncSetAttribute((const void*)msa_core,
                              hipFuncAttributeMaxDynamicSharedMemorySize, (int)SMEM);
    msa_core<<<N_RES, 256, SMEM, stream>>>(
        xn, WqT, WkT, WvT, WgT, WoT, msa_mask, gating_b, output_b, out);
}
